// DotProductAttention_91156385890820
// MI455X (gfx1250) — matrix operand order repaired and checked by reading, not compiled, not run on hardware
//
#include <hip/hip_runtime.h>
#include <math.h>

typedef __attribute__((ext_vector_type(16))) __bf16 v16bf;
typedef __attribute__((ext_vector_type(8))) float v8f;
typedef __attribute__((ext_vector_type(4))) int v4i;

union Frag {
  v16bf v;
  uint4 q[2];
};

// ---- CDNA5 async global->LDS staging (zero VGPR cost), guarded ----
#if defined(__has_builtin)
#if __has_builtin(__builtin_amdgcn_global_load_async_to_lds_b128)
#define USE_ASYNC_LDS 1
#endif
#endif
#ifndef USE_ASYNC_LDS
#define USE_ASYNC_LDS 0
#endif

#if USE_ASYNC_LDS
// param 0 is v4i* per hipcc diagnostic; AS1/AS3 qualify (implicitly convert
// to generic if the builtin's params are generic).
__device__ __forceinline__ void async_copy16(const unsigned short* g,
                                             unsigned short* l) {
  auto gp = (__attribute__((address_space(1))) v4i*)(unsigned long long)g;
  auto lp = (__attribute__((address_space(3))) v4i*)(unsigned)(unsigned long long)l;
  __builtin_amdgcn_global_load_async_to_lds_b128(gp, lp, 0, 0);
}
#endif
__device__ __forceinline__ void wait_async0() {
#if USE_ASYNC_LDS
#if __has_builtin(__builtin_amdgcn_s_wait_asynccnt)
  __builtin_amdgcn_s_wait_asynccnt(0);
#else
  asm volatile("s_wait_asynccnt 0x0" ::: "memory");
#endif
#endif
}

// ---- f32 -> bf16 (RNE) helpers, pure 32-bit integer ops ----
__device__ __forceinline__ unsigned bf16_rnd_bits(float v) {
  unsigned u = __float_as_uint(v);
  return u + 0x7FFFu + ((u >> 16) & 1u);  // bf16 = bits[31:16]
}
__device__ __forceinline__ unsigned short bf16_hi16(float v) {
  return (unsigned short)(bf16_rnd_bits(v) >> 16);
}
__device__ __forceinline__ float bf16_as_f32(float v) {
  return __uint_as_float(bf16_rnd_bits(v) & 0xFFFF0000u);
}
__device__ __forceinline__ unsigned pack_bf16x2(float a, float b) {
  return (bf16_rnd_bits(a) >> 16) | (bf16_rnd_bits(b) & 0xFFFF0000u);
}

// ---------------------------------------------------------------------------
// Elementwise split: src f32 -> (hi, lo) bf16 so src ~= hi + lo.  4 elem/thr.
// ---------------------------------------------------------------------------
__global__ __launch_bounds__(256) void split_bf16_kernel(
    const float* __restrict__ src, unsigned short* __restrict__ h,
    unsigned short* __restrict__ l) {
  int i = blockIdx.x * 256 + threadIdx.x;
  float4 v = ((const float4*)src)[i];
  float h0 = bf16_as_f32(v.x), h1 = bf16_as_f32(v.y);
  float h2 = bf16_as_f32(v.z), h3 = bf16_as_f32(v.w);
  uint2 hp, lp;
  hp.x = pack_bf16x2(v.x, v.y);
  hp.y = pack_bf16x2(v.z, v.w);
  lp.x = pack_bf16x2(v.x - h0, v.y - h1);
  lp.y = pack_bf16x2(v.z - h2, v.w - h3);
  ((uint2*)h)[i] = hp;
  ((uint2*)l)[i] = lp;
}

// ---------------------------------------------------------------------------
// Split + transpose: W f32 [K x N] row-major -> th/tl bf16 [N x K] (B^T form).
// ---------------------------------------------------------------------------
__global__ __launch_bounds__(256) void split_transpose_kernel(
    const float* __restrict__ W, unsigned short* __restrict__ th,
    unsigned short* __restrict__ tl, int K, int N) {
  __shared__ float Ts[32][33];
  const int kb = blockIdx.y * 32, nb = blockIdx.x * 32;
  const int t = threadIdx.x;
  {
    int r = t >> 3, c4 = t & 7;
    float4 v = *(const float4*)(W + (size_t)(kb + r) * N + nb + c4 * 4);
    Ts[r][c4 * 4 + 0] = v.x;
    Ts[r][c4 * 4 + 1] = v.y;
    Ts[r][c4 * 4 + 2] = v.z;
    Ts[r][c4 * 4 + 3] = v.w;
  }
  __syncthreads();
  {
    int n = t >> 3, k4 = t & 7;
    float a0 = Ts[k4 * 4 + 0][n], a1 = Ts[k4 * 4 + 1][n];
    float a2 = Ts[k4 * 4 + 2][n], a3 = Ts[k4 * 4 + 3][n];
    uint2 hp, lp;
    hp.x = pack_bf16x2(a0, a1);
    hp.y = pack_bf16x2(a2, a3);
    lp.x = pack_bf16x2(a0 - bf16_as_f32(a0), a1 - bf16_as_f32(a1));
    lp.y = pack_bf16x2(a2 - bf16_as_f32(a2), a3 - bf16_as_f32(a3));
    size_t o = (size_t)(nb + n) * K + kb + k4 * 4;
    *(uint2*)(th + o) = hp;
    *(uint2*)(tl + o) = lp;
  }
}

// ---------------------------------------------------------------------------
// Split-bf16 GEMM, all operands pre-split bf16:  C = alpha * A x B^T(stored)
//   acc += Ah Bh + Ah Bl + Al Bh   (fp32-grade accuracy on the bf16 WMMA path)
// A: [M x Kd] row-major (hi/lo).  B: [N x Kd] n-major (hi/lo), row stride ldb.
//  MODE 0: plain   MODE 1: skip tiles above diagonal (scores)
//  MODE 2: k-loop truncated at row-tile diagonal (prob @ V)
//  EPI  0: C f32 (ldc)   EPI 1: split bf16 Ch/Cl (ldc)   EPI 2: EPI1 transposed
// 256 thr = 8 waves; 128x64 C tile; wave owns 16x64; K staged 32/turn through
// double-buffered LDS. Staging: GLOBAL_LOAD_ASYNC_TO_LDS_B128 when available
// (issued before compute, zero VGPRs, ASYNCcnt waited); else short-live-range
// reg copies issued after compute (no spills).
// ---------------------------------------------------------------------------
template <int MODE, int EPI>
__global__ __launch_bounds__(256) void gemm_bf16x3(
    const unsigned short* __restrict__ Ahg, const unsigned short* __restrict__ Alg,
    const unsigned short* __restrict__ Bhg, const unsigned short* __restrict__ Blg,
    float* __restrict__ C, unsigned short* __restrict__ Ch,
    unsigned short* __restrict__ Cl, int Kd, int ldb, int ldc, float alpha) {
  const int row0 = blockIdx.y * 128;
  const int col0 = blockIdx.x * 64;
  if (MODE == 1 && col0 > row0 + 127) return;
  const int kmax = (MODE == 2) ? (row0 + 128) : Kd;
  const int nstage = kmax >> 5;

  // rows padded to 40 shorts (80 B): 16 B-aligned b128 fragment loads
  __shared__ alignas(16) unsigned short Ash[2][128][40];
  __shared__ alignas(16) unsigned short Asl[2][128][40];
  __shared__ alignas(16) unsigned short Bsh[2][64][40];  // [n][k]
  __shared__ alignas(16) unsigned short Bsl[2][64][40];

  const int tid  = threadIdx.x;
  const int wave = tid >> 5;
  const int lane = tid & 31;
  const int ml   = lane & 15;
  const int hi   = lane >> 4;
  const int arow = 16 * wave + ml;

  auto stage_to_lds = [&](int kb, int buf) {
#if USE_ASYNC_LDS
#pragma unroll
    for (int i = 0; i < 2; ++i) {
      int f = tid + 256 * i, r = f >> 2, c8 = f & 3;  // 8 shorts per b128
      size_t o = (size_t)(row0 + r) * Kd + kb + c8 * 8;
      async_copy16(Ahg + o, &Ash[buf][r][c8 * 8]);
      async_copy16(Alg + o, &Asl[buf][r][c8 * 8]);
    }
    {
      int r = tid >> 2, c8 = tid & 3;
      size_t o = (size_t)(col0 + r) * ldb + kb + c8 * 8;
      async_copy16(Bhg + o, &Bsh[buf][r][c8 * 8]);
      async_copy16(Blg + o, &Bsl[buf][r][c8 * 8]);
    }
#else
    uint4 rah[2], ral[2], rbh, rbl;
#pragma unroll
    for (int i = 0; i < 2; ++i) {
      int f = tid + 256 * i, r = f >> 2, c8 = f & 3;
      size_t o = (size_t)(row0 + r) * Kd + kb + c8 * 8;
      rah[i] = *(const uint4*)(Ahg + o);
      ral[i] = *(const uint4*)(Alg + o);
    }
    {
      int r = tid >> 2, c8 = tid & 3;
      size_t o = (size_t)(col0 + r) * ldb + kb + c8 * 8;
      rbh = *(const uint4*)(Bhg + o);
      rbl = *(const uint4*)(Blg + o);
    }
#pragma unroll
    for (int i = 0; i < 2; ++i) {
      int f = tid + 256 * i, r = f >> 2, c8 = f & 3;
      *(uint4*)&Ash[buf][r][c8 * 8] = rah[i];
      *(uint4*)&Asl[buf][r][c8 * 8] = ral[i];
    }
    int r = tid >> 2, c8 = tid & 3;
    *(uint4*)&Bsh[buf][r][c8 * 8] = rbh;
    *(uint4*)&Bsl[buf][r][c8 * 8] = rbl;
#endif
  };

  v8f acc[4] = {};

  stage_to_lds(0, 0);
  wait_async0();
  for (int s = 0; s < nstage; ++s) {
    __syncthreads();
    const int buf = s & 1;
#if USE_ASYNC_LDS
    // issue next stage before compute: overlaps WMMA, costs no VGPRs
    if (s + 1 < nstage) stage_to_lds((s + 1) << 5, (s + 1) & 1);
#endif

    // A frag (16-bit A 16x32): lane reads K[8hi..+7], K[16+8hi..+7]
    Frag fah, fal;
    fah.q[0] = *(const uint4*)&Ash[buf][arow][8 * hi];
    fah.q[1] = *(const uint4*)&Ash[buf][arow][16 + 8 * hi];
    fal.q[0] = *(const uint4*)&Asl[buf][arow][8 * hi];
    fal.q[1] = *(const uint4*)&Asl[buf][arow][16 + 8 * hi];

#pragma unroll
    for (int t = 0; t < 4; ++t) {
      const int bn = 16 * t + ml;  // B frag (32x16): K[8hi..+7], K[16+8hi..+7] of col n
      Frag fbh, fbl;
      fbh.q[0] = *(const uint4*)&Bsh[buf][bn][8 * hi];
      fbh.q[1] = *(const uint4*)&Bsh[buf][bn][16 + 8 * hi];
      fbl.q[0] = *(const uint4*)&Bsl[buf][bn][8 * hi];
      fbl.q[1] = *(const uint4*)&Bsl[buf][bn][16 + 8 * hi];
      acc[t] = __builtin_amdgcn_wmma_f32_16x16x32_bf16(
          false, fah.v, false, fbh.v, (short)0, acc[t], false, false);
      acc[t] = __builtin_amdgcn_wmma_f32_16x16x32_bf16(
          false, fah.v, false, fbl.v, (short)0, acc[t], false, false);
      acc[t] = __builtin_amdgcn_wmma_f32_16x16x32_bf16(
          false, fal.v, false, fbh.v, (short)0, acc[t], false, false);
    }

    if (s + 1 < nstage) {
#if !USE_ASYNC_LDS
      // fallback: load+store adjacent (short live range -> no scratch spills)
      stage_to_lds((s + 1) << 5, (s + 1) & 1);
#endif
      wait_async0();
    }
  }

#pragma unroll
  for (int t = 0; t < 4; ++t) {
#pragma unroll
    for (int r = 0; r < 8; ++r) {
      int grow = row0 + 16 * wave + r + 8 * hi;
      int gcol = col0 + 16 * t + ml;
      float v = alpha * acc[t][r];
      if (EPI == 0) {
        C[(size_t)grow * ldc + gcol] = v;
      } else if (EPI == 1) {
        size_t o = (size_t)grow * ldc + gcol;
        Ch[o] = bf16_hi16(v);
        Cl[o] = bf16_hi16(v - bf16_as_f32(v));
      } else {  // transposed split store (Vx^T)
        size_t o = (size_t)gcol * ldc + grow;
        Ch[o] = bf16_hi16(v);
        Cl[o] = bf16_hi16(v - bf16_as_f32(v));
      }
    }
  }
}

// ---------------------------------------------------------------------------
// Causal softmax: reads f32 scores row (cached in LDS), writes SPLIT bf16
// probs; cols > r written as 0 so the PV GEMM needs no masking.
// ---------------------------------------------------------------------------
__global__ __launch_bounds__(256) void softmax_causal_split(
    const float* __restrict__ S, unsigned short* __restrict__ Ph,
    unsigned short* __restrict__ Pl, int n) {
  __shared__ float buf[4096];
  __shared__ float red[256];
  const int r = blockIdx.x;
  const float* row = S + (size_t)r * n;
  const int t = threadIdx.x;

  float m = -INFINITY;
  for (int j = t; j <= r; j += 256) {
    float v = row[j];
    buf[j] = v;
    m = fmaxf(m, v);
  }
  red[t] = m;
  __syncthreads();
  for (int s = 128; s > 0; s >>= 1) {
    if (t < s) red[t] = fmaxf(red[t], red[t + s]);
    __syncthreads();
  }
  m = red[0];
  __syncthreads();

  float sum = 0.0f;
  for (int j = t; j <= r; j += 256) sum += __expf(buf[j] - m);
  red[t] = sum;
  __syncthreads();
  for (int s = 128; s > 0; s >>= 1) {
    if (t < s) red[t] += red[t + s];
    __syncthreads();
  }
  const float inv = 1.0f / red[0];

  for (int j = t; j < n; j += 256) {
    float p = (j <= r) ? __expf(buf[j] - m) * inv : 0.0f;
    size_t o = (size_t)r * n + j;
    Ph[o] = bf16_hi16(p);
    Pl[o] = bf16_hi16(p - bf16_as_f32(p));
  }
}

// ---------------------------------------------------------------------------
extern "C" void kernel_launch(void* const* d_in, const int* in_sizes, int n_in,
                              void* d_out, int out_size, void* d_ws,
                              size_t ws_size, hipStream_t stream) {
  (void)in_sizes; (void)n_in; (void)out_size; (void)ws_size;
  constexpr int BATCH = 4, SEQ = 4096, D = 1024;
  constexpr int MR = BATCH * SEQ;  // 16384 tokens
  const float* x = (const float*)d_in[0];
  const float* Q = (const float*)d_in[1];
  const float* K = (const float*)d_in[2];
  const float* V = (const float*)d_in[3];
  float* out = (float*)d_out;

  // -------- workspace map (332 MB) --------
  const size_t MB = 1024ull * 1024ull;
  char* w = (char*)d_ws;
  unsigned short* xh  = (unsigned short*)(w + 0 * MB);    // 32 MB (reused: Ph)
  unsigned short* xl  = (unsigned short*)(w + 32 * MB);   // 32 MB (reused: Pl)
  unsigned short* Qth = (unsigned short*)(w + 64 * MB);
  unsigned short* Qtl = (unsigned short*)(w + 66 * MB);
  unsigned short* Kth = (unsigned short*)(w + 68 * MB);
  unsigned short* Ktl = (unsigned short*)(w + 70 * MB);
  unsigned short* Vth = (unsigned short*)(w + 72 * MB);
  unsigned short* Vtl = (unsigned short*)(w + 74 * MB);
  unsigned short* Qxh = (unsigned short*)(w + 76 * MB);
  unsigned short* Qxl = (unsigned short*)(w + 108 * MB);
  unsigned short* Kxh = (unsigned short*)(w + 140 * MB);
  unsigned short* Kxl = (unsigned short*)(w + 172 * MB);
  unsigned short* Vxth = (unsigned short*)(w + 204 * MB); // [D][MR] transposed
  unsigned short* Vxtl = (unsigned short*)(w + 236 * MB);
  float* P = (float*)(w + 268 * MB);                      // 64 MB scores
  unsigned short* Ph = xh;  // probs split reuses x split (dead after proj)
  unsigned short* Pl = xl;

  dim3 blk(256);
  const float rsqrt_d = 1.0f / 32.0f;  // 1/sqrt(1024)

  // 1) split x once; split+transpose weights once
  split_bf16_kernel<<<(MR * D / 4) / 256, blk, 0, stream>>>(x, xh, xl);
  dim3 gt(D / 32, D / 32);
  split_transpose_kernel<<<gt, blk, 0, stream>>>(Q, Qth, Qtl, D, D);
  split_transpose_kernel<<<gt, blk, 0, stream>>>(K, Kth, Ktl, D, D);
  split_transpose_kernel<<<gt, blk, 0, stream>>>(V, Vth, Vtl, D, D);

  // 2) projections -> split bf16 outputs (V stored transposed [D][MR])
  dim3 gp(D / 64, MR / 128);
  gemm_bf16x3<0, 1><<<gp, blk, 0, stream>>>(xh, xl, Qth, Qtl, nullptr, Qxh, Qxl,
                                            D, D, D, rsqrt_d);
  gemm_bf16x3<0, 1><<<gp, blk, 0, stream>>>(xh, xl, Kth, Ktl, nullptr, Kxh, Kxl,
                                            D, D, D, 1.0f);
  gemm_bf16x3<0, 2><<<gp, blk, 0, stream>>>(xh, xl, Vth, Vtl, nullptr, Vxth,
                                            Vxtl, D, D, MR, 1.0f);

  // 3) per-batch attention
  for (int b = 0; b < BATCH; ++b) {
    const size_t bo = (size_t)b * SEQ * D;
    float* Ob = out + bo;

    // scores: P = Qx Kx^T (Kx rows already n-major for this product)
    dim3 gs(SEQ / 64, SEQ / 128);
    gemm_bf16x3<1, 0><<<gs, blk, 0, stream>>>(Qxh + bo, Qxl + bo, Kxh + bo,
                                              Kxl + bo, P, nullptr, nullptr, D,
                                              D, SEQ, 1.0f);

    // causal softmax -> split bf16 probs (zeros above diagonal)
    softmax_causal_split<<<SEQ, blk, 0, stream>>>(P, Ph, Pl, SEQ);

    // out = P Vx   (B = Vx^T slab for this batch, row stride MR)
    dim3 gv(D / 64, SEQ / 128);
    gemm_bf16x3<2, 0><<<gv, blk, 0, stream>>>(
        Ph, Pl, Vxth + (size_t)b * SEQ, Vxtl + (size_t)b * SEQ, Ob, nullptr,
        nullptr, SEQ, MR, D, 1.0f);
  }
}
